// GraphAttentionLayer_24318104830455
// MI455X (gfx1250) — compile-verified, matrix operand order checked by reading, not run on hardware
//
#include <hip/hip_runtime.h>
#include <hip/hip_bf16.h>
#include <stdint.h>

#define N_NODES 8192
#define IN_F    512
#define OUT_F   8192
#define ALPHA   0.2f
#define NEG_INF -9000000000000000.0f

typedef __bf16 bf16;
typedef __attribute__((ext_vector_type(16))) __bf16 v16bf;
typedef __attribute__((ext_vector_type(8)))  float  v8f;

union BfPair { unsigned u; bf16 h[2]; };
union Frag   { uint4 q[2]; v16bf v; };

// ---------------------------------------------------------------------------
// CDNA5 async global->LDS copy (ASYNCcnt-tracked). Each active lane moves 16B,
// so one instruction per wave moves 512B. LDS address = workgroup-relative
// byte offset (low 32 bits of the generic shared pointer).
// ---------------------------------------------------------------------------
__device__ __forceinline__ void async_copy_b128(unsigned lds_addr, const bf16* gaddr) {
    asm volatile("global_load_async_to_lds_b128 %0, %1, off"
                 :: "v"(lds_addr), "v"(gaddr)
                 : "memory");
}
__device__ __forceinline__ void wait_asynccnt_4() {
    asm volatile("s_wait_asynccnt 0x4" ::: "memory");
}
__device__ __forceinline__ void wait_asynccnt_0() {
    asm volatile("s_wait_asynccnt 0x0" ::: "memory");
}

// Fragment (16 bf16 = 8 VGPRs) from an LDS row of 32 K-contiguous bf16,
// per the CDNA5 16-bit 16x32 layout: lanes 0-15 hold K={0..7,16..23},
// lanes 16-31 hold K={8..15,24..31}. Exactly two contiguous 16B chunks:
// [base + half*8 .. +8) and [base + 16 + half*8 .. +8)  (bf16 units)
// -> forces 2x ds_load_b128 per fragment.
__device__ __forceinline__ v16bf load_frag_lds(const bf16* base, int half) {
    Frag f;
    f.q[0] = *reinterpret_cast<const uint4*>(base + half * 8);
    f.q[1] = *reinterpret_cast<const uint4*>(base + 16 + half * 8);
    return f.v;
}

// ---------------------------------------------------------------------------
// Generic WMMA GEMM:  C(MxN f32-acc) = A(MxK) * Bsrc(NxK)^T, all bf16 in.
// A rows and Bsrc rows are K-contiguous -> pure b128 async staging.
// 256 threads = 8 waves; 128x128 block tile; K-step 32.
// Triple-buffered LDS (3 x 16KB), single barrier per K-step, prefetch
// distance 2 tiles via ASYNCcnt. Wave (wr=w&3, wc=w>>2) computes 32x64:
// 6 fragment preloads (12 ds_load_b128) then 8 back-to-back WMMAs.
// ---------------------------------------------------------------------------
template <typename OutT>
__global__ __launch_bounds__(256) void wmma_gemm_nt(const bf16* __restrict__ A, size_t lda,
                                                    const bf16* __restrict__ B, size_t ldb,
                                                    OutT* __restrict__ C, size_t ldc,
                                                    int K) {
    // layout per buffer: As 128x32 (8KB) | Bs 128x32 (8KB); 3 buffers = 48KB
    __shared__ __align__(16) bf16 smem[3 * 8192];

    const int tid  = threadIdx.x;
    const int lane = tid & 31;
    const int wave = tid >> 5;
    const int half = lane >> 4;
    const int ln16 = lane & 15;
    const int wr   = wave & 3;   // 0..3 -> 32-row band
    const int wc   = wave >> 2;  // 0..1 -> 64-col band
    const int row0 = blockIdx.y * 128;
    const int col0 = blockIdx.x * 128;

    const unsigned smem_lo = (unsigned)(size_t)smem;   // LDS byte offset of buffer 0

    const v8f zero = {0.f, 0.f, 0.f, 0.f, 0.f, 0.f, 0.f, 0.f};
    v8f acc[2][4];
#pragma unroll
    for (int f = 0; f < 2; ++f)
#pragma unroll
        for (int t = 0; t < 4; ++t) acc[f][t] = zero;

    // Stage one K-tile: A 128x32 + B 128x32 bf16 = 1024 x 16B transfers;
    // 2 A + 2 B async instructions per thread (4 per wave -> ASYNCcnt += 4).
    auto stage = [&](int p, int kb) {
        const unsigned ldsA = smem_lo + (unsigned)(p * 16384);
        const unsigned ldsB = ldsA + 8192u;
#pragma unroll
        for (int it = 0; it < 2; ++it) {
            const int idx = tid + it * 256;
            const int r   = idx >> 2;
            const int seg = (idx & 3) << 3;            // bf16 elems: 0,8,16,24
            async_copy_b128(ldsA + (unsigned)(r * 64 + seg * 2),
                            A + (size_t)(row0 + r) * lda + kb + seg);
        }
#pragma unroll
        for (int it = 0; it < 2; ++it) {
            const int idx = tid + it * 256;
            const int r   = idx >> 2;
            const int seg = (idx & 3) << 3;
            async_copy_b128(ldsB + (unsigned)(r * 64 + seg * 2),
                            B + (size_t)(col0 + r) * ldb + kb + seg);
        }
    };

    const int nT = K / 32;            // number of K tiles (>= 2 for our shapes)
    stage(0, 0);
    if (1 < nT) stage(1, 32);

    for (int i = 0; i < nT; ++i) {
        // In-order async completion: with tiles i and i+1 in flight (8 ops),
        // waiting to <=4 retires tile i. Final tile waits to 0.
        if (i + 1 < nT) wait_asynccnt_4();
        else            wait_asynccnt_0();
        // Single barrier: proves (a) every wave's tile-i copies landed and
        // (b) every wave finished reading buffer (i+2)%3 at iteration i-1,
        // so it is safe to re-stage below.
        __syncthreads();

        const bf16* As = smem + (i % 3) * 8192;
        const bf16* Bs = As + 4096;

        const v16bf af0 = load_frag_lds(As + (wr * 32 + ln16) * 32, half);
        const v16bf af1 = load_frag_lds(As + (wr * 32 + 16 + ln16) * 32, half);
        const v16bf bf0 = load_frag_lds(Bs + (wc * 64 + 0  + ln16) * 32, half);
        const v16bf bf1 = load_frag_lds(Bs + (wc * 64 + 16 + ln16) * 32, half);
        const v16bf bf2 = load_frag_lds(Bs + (wc * 64 + 32 + ln16) * 32, half);
        const v16bf bf3 = load_frag_lds(Bs + (wc * 64 + 48 + ln16) * 32, half);

        if (i + 2 < nT) stage((i + 2) % 3, (i + 2) * 32);   // async, overlaps WMMAs

        acc[0][0] = __builtin_amdgcn_wmma_f32_16x16x32_bf16(false, af0, false, bf0, (short)0, acc[0][0], false, false);
        acc[1][0] = __builtin_amdgcn_wmma_f32_16x16x32_bf16(false, af1, false, bf0, (short)0, acc[1][0], false, false);
        acc[0][1] = __builtin_amdgcn_wmma_f32_16x16x32_bf16(false, af0, false, bf1, (short)0, acc[0][1], false, false);
        acc[1][1] = __builtin_amdgcn_wmma_f32_16x16x32_bf16(false, af1, false, bf1, (short)0, acc[1][1], false, false);
        acc[0][2] = __builtin_amdgcn_wmma_f32_16x16x32_bf16(false, af0, false, bf2, (short)0, acc[0][2], false, false);
        acc[1][2] = __builtin_amdgcn_wmma_f32_16x16x32_bf16(false, af1, false, bf2, (short)0, acc[1][2], false, false);
        acc[0][3] = __builtin_amdgcn_wmma_f32_16x16x32_bf16(false, af0, false, bf3, (short)0, acc[0][3], false, false);
        acc[1][3] = __builtin_amdgcn_wmma_f32_16x16x32_bf16(false, af1, false, bf3, (short)0, acc[1][3], false, false);
    }

    // C/D layout: VGPR r -> M = r + 8*half, N = lane%16.
#pragma unroll
    for (int f = 0; f < 2; ++f)
#pragma unroll
        for (int t = 0; t < 4; ++t)
#pragma unroll
            for (int r = 0; r < 8; ++r) {
                const int m = row0 + wr * 32 + f * 16 + r + 8 * half;
                const int n = col0 + wc * 64 + t * 16 + ln16;
                C[(size_t)m * ldc + n] = (OutT)acc[f][t][r];
            }
}

// ---------------------------------------------------------------------------
// f32 -> bf16 bulk convert (x and W pre-conversion; n % 4 == 0).
// ---------------------------------------------------------------------------
__global__ __launch_bounds__(256) void cvt_f32_bf16(const float* __restrict__ s,
                                                    bf16* __restrict__ d, int n) {
    const int i = (blockIdx.x * 256 + threadIdx.x) * 4;
    if (i < n) {
        const float4 f = *reinterpret_cast<const float4*>(s + i);
        bf16* o = d + i;
        o[0] = (bf16)f.x; o[1] = (bf16)f.y; o[2] = (bf16)f.z; o[3] = (bf16)f.w;
    }
}

// ---------------------------------------------------------------------------
// LDS-tiled bf16 transpose: hT[n][i] = h[i][n]  (coalesced both sides).
// ---------------------------------------------------------------------------
__global__ __launch_bounds__(256) void transpose_bf16(const bf16* __restrict__ src,
                                                      bf16* __restrict__ dst) {
    __shared__ bf16 t[64][68];   // padded to dodge bank conflicts
    const int tid = threadIdx.x;
    const int r0 = blockIdx.y * 64;
    const int c0 = blockIdx.x * 64;

    for (int v = tid; v < 1024; v += 256) {
        const int r = v >> 4, c = (v & 15) << 2;
        const uint2 q =
            *reinterpret_cast<const uint2*>(src + (size_t)(r0 + r) * N_NODES + c0 + c);
        BfPair p0, p1; p0.u = q.x; p1.u = q.y;
        t[c + 0][r] = p0.h[0];
        t[c + 1][r] = p0.h[1];
        t[c + 2][r] = p1.h[0];
        t[c + 3][r] = p1.h[1];
    }
    __syncthreads();
    for (int v = tid; v < 1024; v += 256) {
        const int rr = v >> 4, cc = (v & 15) << 2;
        BfPair p0, p1;
        p0.h[0] = t[rr][cc + 0]; p0.h[1] = t[rr][cc + 1];
        p1.h[0] = t[rr][cc + 2]; p1.h[1] = t[rr][cc + 3];
        uint2 q; q.x = p0.u; q.y = p1.u;
        *reinterpret_cast<uint2*>(dst + (size_t)(c0 + rr) * N_NODES + r0 + cc) = q;
    }
}

// ---------------------------------------------------------------------------
// invcn[j] = 1 / ||h[:, j]||  (coalesced across consecutive columns)
// ---------------------------------------------------------------------------
__global__ __launch_bounds__(256) void gat_colnorm(const bf16* __restrict__ h,
                                                   float* __restrict__ invcn) {
    const int j = blockIdx.x * 256 + threadIdx.x;
    float s = 0.f;
    for (int i = 0; i < N_NODES; ++i) {
        const float v = (float)h[(size_t)i * OUT_F + j];
        s = fmaf(v, v, s);
    }
    invcn[j] = rsqrtf(s);
}

// ---------------------------------------------------------------------------
// Fused scores + mask + row softmax -> attention (bf16).
// One 256-thread block per row i; 32KB score row held in LDS.
// hT makes the "h[j,i]" term a coalesced row read.
// ---------------------------------------------------------------------------
__global__ __launch_bounds__(256) void gat_attn_softmax(const bf16* __restrict__ h,
                                                        const bf16* __restrict__ hT,
                                                        const int* __restrict__ adj,
                                                        const float* __restrict__ a,
                                                        const float* __restrict__ invcn,
                                                        bf16* __restrict__ attn) {
    __shared__ float erow[N_NODES];
    __shared__ float red[256];
    const int i   = blockIdx.x;
    const int tid = threadIdx.x;

    const float a1i   = a[i];
    const float inv_i = invcn[i];

    float mx = NEG_INF;
    for (int j = tid; j < N_NODES; j += 256) {
        const float hij = (float)h[(size_t)i * OUT_F + j];
        const float hji = (float)hT[(size_t)i * N_NODES + j];
        const float t   = a1i * hij * invcn[j] + inv_i * a[OUT_F + j] * hji;
        const float e   = (t > 0.f) ? t : ALPHA * t;
        const float s   = (adj[(size_t)i * N_NODES + j] > 0) ? e : NEG_INF;
        erow[j] = s;
        mx = fmaxf(mx, s);
    }
    red[tid] = mx;
    __syncthreads();
    for (int off = 128; off > 0; off >>= 1) {
        if (tid < off) red[tid] = fmaxf(red[tid], red[tid + off]);
        __syncthreads();
    }
    mx = red[0];
    __syncthreads();

    float sum = 0.f;
    for (int j = tid; j < N_NODES; j += 256) {
        const float ex = __expf(erow[j] - mx);
        erow[j] = ex;
        sum += ex;
    }
    red[tid] = sum;
    __syncthreads();
    for (int off = 128; off > 0; off >>= 1) {
        if (tid < off) red[tid] += red[tid + off];
        __syncthreads();
    }
    const float inv_sum = 1.0f / red[0];
    for (int j = tid; j < N_NODES; j += 256)
        attn[(size_t)i * N_NODES + j] = (bf16)(erow[j] * inv_sum);
}

// ---------------------------------------------------------------------------
extern "C" void kernel_launch(void* const* d_in, const int* in_sizes, int n_in,
                              void* d_out, int out_size, void* d_ws, size_t ws_size,
                              hipStream_t stream) {
    (void)in_sizes; (void)n_in; (void)out_size; (void)ws_size;

    const float* x   = (const float*)d_in[0];   // (N, IN)    f32
    const int*   adj = (const int*)  d_in[1];   // (N, N)     i32
    const float* W   = (const float*)d_in[2];   // (OUT, IN)  f32
    const float* a   = (const float*)d_in[3];   // (2*OUT, 1) f32
    float* out = (float*)d_out;                 // (N, OUT)   f32

    char* ws = (char*)d_ws;
    size_t off = 0;
    bf16* h    = (bf16*)(ws + off); off += (size_t)N_NODES * OUT_F * sizeof(bf16);   // 128 MB
    bf16* hT   = (bf16*)(ws + off); off += (size_t)N_NODES * OUT_F * sizeof(bf16);   // 128 MB
    bf16* attn = (bf16*)(ws + off); off += (size_t)N_NODES * N_NODES * sizeof(bf16); // 128 MB
    bf16* xb   = (bf16*)(ws + off); off += (size_t)N_NODES * IN_F * sizeof(bf16);    //   8 MB
    bf16* Wb   = (bf16*)(ws + off); off += (size_t)OUT_F * IN_F * sizeof(bf16);      //   8 MB
    float* invcn = (float*)(ws + off);                                               //  32 KB

    // 1) f32 -> bf16 operand conversion
    cvt_f32_bf16<<<(N_NODES * IN_F / 4 + 255) / 256, 256, 0, stream>>>(x, xb, N_NODES * IN_F);
    cvt_f32_bf16<<<(OUT_F * IN_F / 4 + 255) / 256, 256, 0, stream>>>(W, Wb, OUT_F * IN_F);

    // 2) h = x @ W^T  (bf16 WMMA, async triple-buffered)
    dim3 gGemm(OUT_F / 128, N_NODES / 128);
    wmma_gemm_nt<bf16><<<gGemm, 256, 0, stream>>>(xb, (size_t)IN_F, Wb, (size_t)IN_F,
                                                  h, (size_t)OUT_F, IN_F);

    // 3) hT = h^T  (makes GEMM2 B-operand and softmax column reads K-contiguous)
    transpose_bf16<<<dim3(OUT_F / 64, N_NODES / 64), 256, 0, stream>>>(h, hT);

    // 4) column norms
    gat_colnorm<<<OUT_F / 256, 256, 0, stream>>>(h, invcn);

    // 5) fused scores + mask + softmax
    gat_attn_softmax<<<N_NODES, 256, 0, stream>>>(h, hT, adj, a, invcn, attn);

    // 6) out = attention @ h  == attention @ (hT)^T  (dominant 1.1 TFLOP GEMM)
    wmma_gemm_nt<float><<<gGemm, 256, 0, stream>>>(attn, (size_t)N_NODES, hT, (size_t)N_NODES,
                                                   out, (size_t)OUT_F, N_NODES);
}
